// TransformerEncoderLayer_88484916232354
// MI455X (gfx1250) — matrix operand order repaired and checked by reading, not compiled, not run on hardware
//
#include <hip/hip_runtime.h>

#define TN 197
#define BTC 128
#define DMODEL 768
#define NHEAD 12
#define HDIM 64
#define TDIM 8
#define BDIM 16
#define BNT (BDIM * TN)        // 3152
#define TOK (TN * BTC)         // 25216
#define FFND 3072
#define AD 192
#define NN (TN * TN)           // 38809
#define BH (BTC * NHEAD)       // 1536

typedef __attribute__((ext_vector_type(8)))  float        v8f_t;
typedef __attribute__((ext_vector_type(16))) __bf16       bf16x16;
typedef __attribute__((ext_vector_type(8)))  __bf16       bf16x8;
typedef __attribute__((ext_vector_type(4)))  unsigned int u32x4;
typedef __attribute__((ext_vector_type(8)))  int          i32x8;
typedef __attribute__((ext_vector_type(4)))  int          i32x4;

union FragU { bf16x16 v; bf16x8 h[2]; };

#define HAS_TDM __has_builtin(__builtin_amdgcn_tensor_load_to_lds)

struct GemmP {
    const __bf16 *A, *B;
    const float *bias, *resid;
    float *C; __bf16 *Cb;
    int M, N, K, transB, outBf;
    long lda, ldb, ldc;
    float alpha;
    int divA, divB, divC;
    long soA, siA, soB, siB, soC, siC;
};

__device__ __forceinline__ float gelu_f(float v) {
    return 0.5f * v * (1.f + erff(v * 0.70710678118f));
}

#if HAS_TDM
// One TDM 2D tile load: rows x cols bf16 from global (strideElems row pitch) into
// LDS at ldsAddr, +4 DWORDs pad every 16 DWORDs (one 32-elem row) -> 80B LDS pitch.
__device__ __forceinline__ void tdm_load_tile(const __bf16* gptr, unsigned ldsAddr,
                                              int rows, int cols, long strideElems) {
    unsigned long ga = (unsigned long)gptr;
    u32x4 g0;
    g0.x = (unsigned)__builtin_amdgcn_readfirstlane(1);                 // count=1
    g0.y = (unsigned)__builtin_amdgcn_readfirstlane((int)ldsAddr);      // lds_addr
    g0.z = (unsigned)__builtin_amdgcn_readfirstlane((int)(unsigned)(ga & 0xFFFFFFFFu));
    g0.w = (unsigned)__builtin_amdgcn_readfirstlane(
               (int)(((unsigned)((ga >> 32) & 0x01FFFFFFu)) | 0x80000000u)); // type=2
    // group1: data_size=2B (1), pad_enable, pad_interval=16DW (3), pad_amount=4DW (3)
    unsigned long q0 = 0x10000ull | 0x100000ull | (3ull << 22) | (3ull << 25)
                     | (((unsigned long)(cols & 0xFFFF)) << 48);        // tensor_dim0 lo16
    unsigned long q1 = ((unsigned long)(((unsigned)cols) >> 16))        // tensor_dim0 hi16
                     | (((unsigned long)(unsigned)rows) << 16)          // tensor_dim1
                     | (32ull << 48);                                   // tile_dim0 = 32
    unsigned long q2 = 16ull                                            // tile_dim1 = 16
                     | (((unsigned long)(strideElems & 0xFFFFFFFFl)) << 32); // stride lo32
    unsigned long q3 = (unsigned long)((strideElems >> 32) & 0xFFFF);   // stride hi16
    i32x8 g1;
    g1[0] = __builtin_amdgcn_readfirstlane((int)(unsigned)q0);
    g1[1] = __builtin_amdgcn_readfirstlane((int)(unsigned)(q0 >> 32));
    g1[2] = __builtin_amdgcn_readfirstlane((int)(unsigned)q1);
    g1[3] = __builtin_amdgcn_readfirstlane((int)(unsigned)(q1 >> 32));
    g1[4] = __builtin_amdgcn_readfirstlane((int)(unsigned)q2);
    g1[5] = __builtin_amdgcn_readfirstlane((int)(unsigned)(q2 >> 32));
    g1[6] = __builtin_amdgcn_readfirstlane((int)(unsigned)q3);
    g1[7] = __builtin_amdgcn_readfirstlane((int)(unsigned)(q3 >> 32));
    i32x4 gz4 = {0, 0, 0, 0};
    i32x8 gz8 = {0, 0, 0, 0, 0, 0, 0, 0};
    __builtin_amdgcn_tensor_load_to_lds(g0, g1, gz4, gz4, gz8, 0);
}
#endif

// ---------------- Generic batched bf16-WMMA GEMM ----------------
// FULL=1: M%128==0, N%128==0, K%32==0, lda/ldb 16B-aligned -> no tail guards.
#define BM 128
#define BNN 128
#define BKK 32
#define BKP 40   // LDS pitch in bf16 (80B)

template <int FULL>
__global__ __launch_bounds__(256) void gemm_wmma(GemmP p) {
    __shared__ __attribute__((aligned(16))) __bf16 As[2][BM][BKP];  // TDM double buffer
    __shared__ __attribute__((aligned(16))) __bf16 Bs[BNN][BKP];

    const int bz = blockIdx.z;
    const __bf16* A  = p.A + (long)(bz / p.divA) * p.soA + (long)(bz % p.divA) * p.siA;
    const __bf16* Bm = p.B + (long)(bz / p.divB) * p.soB + (long)(bz % p.divB) * p.siB;
    const long coff  = (long)(bz / p.divC) * p.soC + (long)(bz % p.divC) * p.siC;

    const int m0 = blockIdx.y * BM, n0 = blockIdx.x * BNN;
    const int tid = threadIdx.x, lane = tid & 31, wave = tid >> 5;
    const int wm = (wave & 3) * 32, wn = (wave >> 2) * 64;

    const v8f_t zero8 = {0.f, 0.f, 0.f, 0.f, 0.f, 0.f, 0.f, 0.f};
    v8f_t acc[2][4];
#pragma unroll
    for (int i = 0; i < 2; i++)
#pragma unroll
        for (int j = 0; j < 4; j++) acc[i][j] = zero8;

    const int alignedA = FULL || ((p.lda & 7) == 0);
    const int alignedB = FULL || ((p.ldb & 7) == 0);
    const int nk = (p.K + BKK - 1) / BKK;

#if HAS_TDM
    auto issueA = [&](int it) {
        int k0 = it * BKK;
        int row0 = m0 + wave * 16;
        const __bf16* gp = A + (long)row0 * p.lda + k0;
        unsigned lds = (unsigned)(unsigned long)&As[it & 1][wave * 16][0];
        int remM, remK;
        if (FULL) { remM = 16; remK = BKK; }
        else {
            remM = p.M - row0; remM = remM < 0 ? 0 : (remM > 16 ? 16 : remM);
            remK = p.K - k0;   remK = remK > BKK ? BKK : remK;
        }
        tdm_load_tile(gp, lds, remM, remK, p.lda);
    };
    issueA(0);   // pipeline prologue
#endif

    for (int it = 0; it < nk; ++it) {
        const int k0 = it * BKK;
        const int ab = it & 1;
        // ---- A tile: async TDM double-buffer (or sync fallback) ----
#if HAS_TDM
        if (it + 1 < nk) issueA(it + 1);
#else
#pragma unroll
        for (int i = 0; i < 2; i++) {
            int id = tid + 256 * i;
            int r = id >> 2, c8 = (id & 3) << 3;
            int gm = m0 + r, gk = k0 + c8;
            if (FULL || (gm < p.M && alignedA && gk + 7 < p.K)) {
                *(bf16x8*)&As[ab][r][c8] = *(const bf16x8*)(A + (long)gm * p.lda + gk);
            } else {
#pragma unroll
                for (int j = 0; j < 8; j++) {
                    __bf16 v = (__bf16)0.f;
                    if (gm < p.M && gk + j < p.K) v = A[(long)gm * p.lda + gk + j];
                    As[ab][r][c8 + j] = v;
                }
            }
        }
#endif
        // ---- B tile staged as Bs[n][k] ----
        if (p.transB) {
#pragma unroll
            for (int i = 0; i < 2; i++) {
                int id = tid + 256 * i;
                int r = id >> 2, c8 = (id & 3) << 3;
                int gn = n0 + r, gk = k0 + c8;
                if (FULL || (gn < p.N && alignedB && gk + 7 < p.K)) {
                    *(bf16x8*)&Bs[r][c8] = *(const bf16x8*)(Bm + (long)gn * p.ldb + gk);
                } else {
#pragma unroll
                    for (int j = 0; j < 8; j++) {
                        __bf16 v = (__bf16)0.f;
                        if (gn < p.N && gk + j < p.K) v = Bm[(long)gn * p.ldb + gk + j];
                        Bs[r][c8 + j] = v;
                    }
                }
            }
        } else {
#pragma unroll
            for (int i = 0; i < 2; i++) {
                int id = tid + 256 * i;
                int kr = id >> 4, c8 = (id & 15) << 3;
                int gk = k0 + kr, gn = n0 + c8;
                __attribute__((aligned(16))) __bf16 tmp[8];
                if (FULL || (gk < p.K && alignedB && gn + 7 < p.N)) {
                    *(bf16x8*)tmp = *(const bf16x8*)(Bm + (long)gk * p.ldb + gn);
                } else {
#pragma unroll
                    for (int j = 0; j < 8; j++) {
                        __bf16 v = (__bf16)0.f;
                        if (gk < p.K && gn + j < p.N) v = Bm[(long)gk * p.ldb + gn + j];
                        tmp[j] = v;
                    }
                }
#pragma unroll
                for (int j = 0; j < 8; j++) Bs[c8 + j][kr] = tmp[j];
            }
        }
        // prefetch next B K-tile into cache
        if (k0 + BKK < p.K) {
            long nkk = k0 + BKK;
            if (p.transB) {
                int r = tid >> 1;
                int gn = n0 + r;
                if (FULL || gn < p.N)
                    __builtin_prefetch(Bm + (long)gn * p.ldb + nkk + (tid & 1) * 16, 0, 1);
            } else {
                int kr = tid >> 3;
                if (FULL || nkk + kr < p.K)
                    __builtin_prefetch(Bm + (nkk + kr) * p.ldb + n0 + (tid & 7) * 16, 0, 1);
            }
        }
#if HAS_TDM
        if (it + 1 < nk) __builtin_amdgcn_s_wait_tensorcnt(1);  // prev tile done, next in flight
        else             __builtin_amdgcn_s_wait_tensorcnt(0);
#endif
        __syncthreads();

        // ---- fragments per ISA 7.12.2 layout ----
        const int mlane = lane & 15;
        const int kbA = (lane >> 4) * 8;
        const int kbB = (lane >> 4) * 8;
        FragU fa[2], fb[4];
#pragma unroll
        for (int i = 0; i < 2; i++) {
            fa[i].h[0] = *(const bf16x8*)&As[ab][wm + i * 16 + mlane][kbA];
            fa[i].h[1] = *(const bf16x8*)&As[ab][wm + i * 16 + mlane][16 + kbA];
        }
#pragma unroll
        for (int j = 0; j < 4; j++) {
            fb[j].h[0] = *(const bf16x8*)&Bs[wn + j * 16 + mlane][kbB];
            fb[j].h[1] = *(const bf16x8*)&Bs[wn + j * 16 + mlane][16 + kbB];
        }
#pragma unroll
        for (int i = 0; i < 2; i++)
#pragma unroll
            for (int j = 0; j < 4; j++)
                acc[i][j] = __builtin_amdgcn_wmma_f32_16x16x32_bf16(
                    false, fa[i].v, false, fb[j].v, (short)0, acc[i][j], false, false);
        __syncthreads();
    }

    const int nlane = lane & 15;
    const int mh = (lane >> 4) * 8;
#pragma unroll
    for (int i = 0; i < 2; i++)
#pragma unroll
        for (int j = 0; j < 4; j++)
#pragma unroll
            for (int r = 0; r < 8; r++) {
                int gm = m0 + wm + i * 16 + mh + r;
                int gn = n0 + wn + j * 16 + nlane;
                if (FULL || (gm < p.M && gn < p.N)) {
                    float v = acc[i][j][r];
                    if (p.bias)  v += p.bias[gn];
                    v *= p.alpha;
                    if (p.resid) v += p.resid[coff + (long)gm * p.ldc + gn];
                    if (p.outBf) p.Cb[coff + (long)gm * p.ldc + gn] = (__bf16)v;
                    else         p.C [coff + (long)gm * p.ldc + gn] = v;
                }
            }
}

// ---------------- LayerNorm (bf16 out; optional temporal-gather permutation) -----
__global__ __launch_bounds__(256) void ln_kernel(const float* in, const float* g,
                                                 const float* b, __bf16* out,
                                                 int W, int perm) {
    __shared__ float red[256];
    int r = blockIdx.x;
    long irow;
    if (perm) {
        int t = r / BNT;
        int rem = r - t * BNT;
        int bb = rem / TN;
        int n = rem - bb * TN;
        irow = (long)(n * BTC + bb * TDIM + t);
    } else {
        irow = r;
    }
    const float* x = in + irow * (long)W;
    int tid = threadIdx.x;
    int nIt = (W + 255) >> 8;
    float vals[12];
    float s = 0.f;
    for (int i = 0; i < nIt; i++) {
        int c = tid + (i << 8);
        float v = (c < W) ? x[c] : 0.f;
        vals[i] = v;
        s += v;
    }
    red[tid] = s; __syncthreads();
    for (int o = 128; o > 0; o >>= 1) { if (tid < o) red[tid] += red[tid + o]; __syncthreads(); }
    float mean = red[0] / (float)W;
    __syncthreads();
    s = 0.f;
    for (int i = 0; i < nIt; i++) {
        int c = tid + (i << 8);
        float d = (c < W) ? (vals[i] - mean) : 0.f;
        s += d * d;
    }
    red[tid] = s; __syncthreads();
    for (int o = 128; o > 0; o >>= 1) { if (tid < o) red[tid] += red[tid + o]; __syncthreads(); }
    float rstd = rsqrtf(red[0] / (float)W + 1e-5f);
    __bf16* o = out + (long)r * W;
    for (int i = 0; i < nIt; i++) {
        int c = tid + (i << 8);
        if (c < W) o[c] = (__bf16)((vals[i] - mean) * rstd * g[c] + b[c]);
    }
}

// ---------------- Tiny temporal attention (seq = T = 8) ----------------
__global__ __launch_bounds__(64) void k_attn_t(const float* q, const float* k,
                                               const float* v, float* o) {
    __shared__ float qs[TDIM][HDIM], ks[TDIM][HDIM], vs[TDIM][HDIM], ss[TDIM][TDIM];
    int bhid = blockIdx.x;
    int bn = bhid / NHEAD, h = bhid - bn * NHEAD;
    int d = threadIdx.x;
    long coff = (long)bn * DMODEL + h * HDIM + d;
    for (int t = 0; t < TDIM; t++) {
        long ro = (long)t * BNT * DMODEL + coff;
        qs[t][d] = q[ro]; ks[t][d] = k[ro]; vs[t][d] = v[ro];
    }
    __syncthreads();
    int ti = d >> 3, tj = d & 7;
    float s = 0.f;
    for (int e = 0; e < HDIM; e++) s += qs[ti][e] * ks[tj][e];
    ss[ti][tj] = s;
    __syncthreads();
    if (d < TDIM) {
        float mx = -3.0e38f;
        for (int j2 = 0; j2 < TDIM; j2++) mx = fmaxf(mx, ss[d][j2]);
        float sum = 0.f;
        for (int j2 = 0; j2 < TDIM; j2++) { float e2 = expf(ss[d][j2] - mx); ss[d][j2] = e2; sum += e2; }
        float inv = 1.f / sum;
        for (int j2 = 0; j2 < TDIM; j2++) ss[d][j2] *= inv;
    }
    __syncthreads();
    for (int t = 0; t < TDIM; t++) {
        float a = 0.f;
        for (int j2 = 0; j2 < TDIM; j2++) a += ss[t][j2] * vs[j2][d];
        o[(long)t * BNT * DMODEL + coff] = a;
    }
}

// ---------------- Row softmax with bias; bf16 probabilities out ----------------
__global__ __launch_bounds__(256) void k_softmax(const float* S, const float* bias,
                                                 __bf16* P) {
    __shared__ float red[256];
    int r = blockIdx.x;
    int bh = r / TN, i = r - bh * TN, h = bh % NHEAD;
    const float* row = S + (long)bh * NN + (long)i * TN;
    const float* br  = bias + ((long)h * TN + i) * TN;
    __bf16* prow = P + (long)bh * NN + (long)i * TN;
    int tid = threadIdx.x;
    float v = -3.0e38f;
    if (tid < TN) v = row[tid] + br[tid];
    red[tid] = v; __syncthreads();
    for (int o = 128; o > 0; o >>= 1) { if (tid < o) red[tid] = fmaxf(red[tid], red[tid + o]); __syncthreads(); }
    float mx = red[0]; __syncthreads();
    float e = 0.f;
    if (tid < TN) e = expf(v - mx);
    red[tid] = e; __syncthreads();
    for (int o = 128; o > 0; o >>= 1) { if (tid < o) red[tid] += red[tid + o]; __syncthreads(); }
    float inv = 1.f / red[0];
    if (tid < TN) prow[tid] = (__bf16)(e * inv);
}

// ---------------- Small elementwise kernels ----------------
__global__ void k_bias_gather(const float* rel, const int* rp, float* out) {
    long idx = (long)blockIdx.x * 256 + threadIdx.x;
    long total = (long)NHEAD * NN;
    if (idx >= total) return;
    int h = (int)(idx / NN);
    int ij = (int)(idx - (long)h * NN);
    out[idx] = rel[(long)rp[ij] * NHEAD + h];
}
__global__ void k_cvt(const float* in, __bf16* out, long n) {
    long i = (long)blockIdx.x * 256 + threadIdx.x;
    if (i < n) out[i] = (__bf16)in[i];
}
__global__ void k_gelu_bf(__bf16* a, long n) {
    long i = (long)blockIdx.x * 256 + threadIdx.x;
    if (i < n) a[i] = (__bf16)gelu_f((float)a[i]);
}
__global__ void k_gelu_mul(float* a, const float* b, long n) {
    long i = (long)blockIdx.x * 256 + threadIdx.x;
    if (i < n) a[i] = gelu_f(a[i]) * b[i];
}
__global__ void k_add_perm(const float* x, const float* y, float* o) {
    long i = (long)blockIdx.x * 256 + threadIdx.x;
    long tot = (long)TOK * DMODEL;
    if (i >= tot) return;
    int r = (int)(i / DMODEL), c = (int)(i - (long)r * DMODEL);
    int n = r / BTC, rem = r - n * BTC, b = rem >> 3, t = rem & 7;
    long yr = (long)(t * BNT + b * TN + n);
    o[i] = x[i] + y[yr * DMODEL + c];
}
__global__ void k_axpg(const float* x, const float* g, const float* h, float* o) {
    long i = (long)blockIdx.x * 256 + threadIdx.x;
    long tot = (long)TOK * DMODEL;
    if (i >= tot) return;
    int c = (int)(i % DMODEL);
    o[i] = x[i] + g[c] * h[i];
}
__global__ void k_final(const float* x, const float* g2, const float* ffn,
                        const float* ma, float* o) {
    long i = (long)blockIdx.x * 256 + threadIdx.x;
    long tot = (long)TOK * DMODEL;
    if (i >= tot) return;
    int c = (int)(i % DMODEL);
    o[i] = x[i] + g2[c] * ffn[i] + 0.5f * ma[i];
}

// ---------------- Host-side orchestration ----------------
static void launch_gemm(hipStream_t st, const __bf16* A, const __bf16* B, const float* bias,
                        const float* resid, float* C, __bf16* Cb, int M, int N, int K,
                        long lda, long ldb, long ldc, int transB, float alpha,
                        int batch = 1,
                        int divA = 1, long soA = 0, long siA = 0,
                        int divB = 1, long soB = 0, long siB = 0,
                        int divC = 1, long soC = 0, long siC = 0) {
    GemmP p;
    p.A = A; p.B = B; p.bias = bias; p.resid = resid; p.C = C; p.Cb = Cb;
    p.M = M; p.N = N; p.K = K; p.transB = transB; p.outBf = (Cb != nullptr);
    p.lda = lda; p.ldb = ldb; p.ldc = ldc; p.alpha = alpha;
    p.divA = divA; p.divB = divB; p.divC = divC;
    p.soA = soA; p.siA = siA; p.soB = soB; p.siB = siB; p.soC = soC; p.siC = siC;
    dim3 grid((N + BNN - 1) / BNN, (M + BM - 1) / BM, batch);
    bool full = (M % BM == 0) && (N % BNN == 0) && (K % BKK == 0) &&
                ((lda & 7) == 0) && ((ldb & 7) == 0);
    if (full) gemm_wmma<1><<<grid, dim3(256), 0, st>>>(p);
    else      gemm_wmma<0><<<grid, dim3(256), 0, st>>>(p);
}

static inline int eblocks(long n) { return (int)((n + 255) / 256); }

extern "C" void kernel_launch(void* const* d_in, const int* in_sizes, int n_in,
                              void* d_out, int out_size, void* d_ws, size_t ws_size,
                              hipStream_t stream) {
    const float* x      = (const float*)d_in[0];
    const int*   rp     = (const int*)  d_in[1];
    const float* ln1_g  = (const float*)d_in[2];
    const float* ln1_b  = (const float*)d_in[3];
    const float* Wq     = (const float*)d_in[4];
    const float* bq     = (const float*)d_in[5];
    const float* Wk     = (const float*)d_in[6];
    const float* Wv     = (const float*)d_in[7];
    const float* bv     = (const float*)d_in[8];
    const float* Wo     = (const float*)d_in[9];
    const float* bo     = (const float*)d_in[10];
    const float* aln_g  = (const float*)d_in[11];
    const float* aln_b  = (const float*)d_in[12];
    const float* rel    = (const float*)d_in[13];
    const float* gamma1 = (const float*)d_in[14];
    const float* gamma2 = (const float*)d_in[15];
    const float* lnf_g  = (const float*)d_in[16];
    const float* lnf_b  = (const float*)d_in[17];
    const float* W0     = (const float*)d_in[18];
    const float* W1     = (const float*)d_in[19];
    const float* fln_g  = (const float*)d_in[20];
    const float* fln_b  = (const float*)d_in[21];
    const float* Wf     = (const float*)d_in[22];
    const float* bfb    = (const float*)d_in[23];
    const float* ta1w   = (const float*)d_in[24];
    const float* ta1b   = (const float*)d_in[25];
    const float* ta2w   = (const float*)d_in[26];
    const float* ta2b   = (const float*)d_in[27];
    const float* sa1w   = (const float*)d_in[28];
    const float* sa1b   = (const float*)d_in[29];
    const float* sa2w   = (const float*)d_in[30];
    const float* sa2b   = (const float*)d_in[31];
    const float* ma1w   = (const float*)d_in[32];
    const float* ma1b   = (const float*)d_in[33];
    const float* ma2w   = (const float*)d_in[34];
    const float* ma2b   = (const float*)d_in[35];

    char* base = (char*)d_ws;
    size_t off = 0;
    auto allocF = [&](long n) { float* p = (float*)(base + off);
                                off += ((size_t)n * 4 + 15) & ~(size_t)15; return p; };
    auto allocB = [&](long n) { __bf16* p = (__bf16*)(base + off);
                                off += ((size_t)n * 2 + 15) & ~(size_t)15; return p; };

    const long TOKD = (long)TOK * DMODEL;
    // fp32 scratch
    float* bA    = allocF(TOKD);
    float* bB    = allocF(TOKD);
    float* bC    = allocF(TOKD);
    float* bD    = allocF(TOKD);
    float* bE    = allocF(TOKD);
    float* bF    = allocF(TOKD);
    float* bBias = allocF((long)NHEAD * NN);
    float* bS    = allocF((long)BH * NN);
    float* bG0   = allocF((long)TOK * FFND);
    float* bG1   = allocF((long)TOK * FFND);
    // bf16 weight copies
    __bf16* cWq  = allocB((long)DMODEL * DMODEL);
    __bf16* cWk  = allocB((long)DMODEL * DMODEL);
    __bf16* cWv  = allocB((long)DMODEL * DMODEL);
    __bf16* cWo  = allocB((long)DMODEL * DMODEL);
    __bf16* cW0  = allocB((long)DMODEL * FFND);
    __bf16* cW1  = allocB((long)DMODEL * FFND);
    __bf16* cWf  = allocB((long)FFND * DMODEL);
    __bf16* cTa1 = allocB((long)DMODEL * AD);
    __bf16* cTa2 = allocB((long)AD * DMODEL);
    __bf16* cSa1 = allocB((long)DMODEL * AD);
    __bf16* cSa2 = allocB((long)AD * DMODEL);
    __bf16* cMa1 = allocB((long)DMODEL * AD);
    __bf16* cMa2 = allocB((long)AD * DMODEL);
    // bf16 activation buffers
    __bf16* bL   = allocB(TOKD);
    __bf16* bX   = allocB(TOKD);
    __bf16* bQb  = allocB(TOKD);
    __bf16* bKb  = allocB(TOKD);
    __bf16* bVb  = allocB(TOKD);
    __bf16* bH0b = allocB(TOKD);
    __bf16* bAdb = allocB((long)TOK * AD);
    __bf16* bPb  = allocB((long)BH * NN);
    __bf16* bGbf = allocB((long)TOK * FFND);

    // convert weights to bf16 once per launch
    k_cvt<<<eblocks((long)DMODEL * DMODEL), 256, 0, stream>>>(Wq, cWq, (long)DMODEL * DMODEL);
    k_cvt<<<eblocks((long)DMODEL * DMODEL), 256, 0, stream>>>(Wk, cWk, (long)DMODEL * DMODEL);
    k_cvt<<<eblocks((long)DMODEL * DMODEL), 256, 0, stream>>>(Wv, cWv, (long)DMODEL * DMODEL);
    k_cvt<<<eblocks((long)DMODEL * DMODEL), 256, 0, stream>>>(Wo, cWo, (long)DMODEL * DMODEL);
    k_cvt<<<eblocks((long)DMODEL * FFND), 256, 0, stream>>>(W0, cW0, (long)DMODEL * FFND);
    k_cvt<<<eblocks((long)DMODEL * FFND), 256, 0, stream>>>(W1, cW1, (long)DMODEL * FFND);
    k_cvt<<<eblocks((long)FFND * DMODEL), 256, 0, stream>>>(Wf, cWf, (long)FFND * DMODEL);
    k_cvt<<<eblocks((long)DMODEL * AD), 256, 0, stream>>>(ta1w, cTa1, (long)DMODEL * AD);
    k_cvt<<<eblocks((long)AD * DMODEL), 256, 0, stream>>>(ta2w, cTa2, (long)AD * DMODEL);
    k_cvt<<<eblocks((long)DMODEL * AD), 256, 0, stream>>>(sa1w, cSa1, (long)DMODEL * AD);
    k_cvt<<<eblocks((long)AD * DMODEL), 256, 0, stream>>>(sa2w, cSa2, (long)AD * DMODEL);
    k_cvt<<<eblocks((long)DMODEL * AD), 256, 0, stream>>>(ma1w, cMa1, (long)DMODEL * AD);
    k_cvt<<<eblocks((long)AD * DMODEL), 256, 0, stream>>>(ma2w, cMa2, (long)AD * DMODEL);

    const long LDQ = (long)BTC * DMODEL;  // 98304
    const float scale = 0.125f;           // 64^-0.5

    // ===== Temporal attention branch =====
    ln_kernel<<<TOK, 256, 0, stream>>>(x, ln1_g, ln1_b, bL, DMODEL, 1);
    launch_gemm(stream, bL, cWq, bq, nullptr, bB, nullptr, TOK, DMODEL, DMODEL, DMODEL, DMODEL, DMODEL, 0, scale);
    launch_gemm(stream, bL, cWk, nullptr, nullptr, bC, nullptr, TOK, DMODEL, DMODEL, DMODEL, DMODEL, DMODEL, 0, 1.f);
    launch_gemm(stream, bL, cWv, bv, nullptr, bD, nullptr, TOK, DMODEL, DMODEL, DMODEL, DMODEL, DMODEL, 0, 1.f);
    k_attn_t<<<BNT * NHEAD, 64, 0, stream>>>(bB, bC, bD, bE);
    ln_kernel<<<TOK, 256, 0, stream>>>(bE, aln_g, aln_b, bL, DMODEL, 0);
    launch_gemm(stream, bL, cWo, bo, nullptr, nullptr, bH0b, TOK, DMODEL, DMODEL, DMODEL, DMODEL, DMODEL, 0, 1.f);
    launch_gemm(stream, bH0b, cTa1, ta1b, nullptr, nullptr, bAdb, TOK, AD, DMODEL, DMODEL, AD, AD, 0, 1.f);
    k_gelu_bf<<<eblocks((long)TOK * AD), 256, 0, stream>>>(bAdb, (long)TOK * AD);
    launch_gemm(stream, bAdb, cTa2, ta2b, nullptr, bB, nullptr, TOK, DMODEL, AD, AD, DMODEL, DMODEL, 0, 1.f);
    k_add_perm<<<eblocks(TOKD), 256, 0, stream>>>(x, bB, bE);   // x1

    // ===== Spatial attention branch =====
    k_bias_gather<<<eblocks((long)NHEAD * NN), 256, 0, stream>>>(rel, rp, bBias);
    ln_kernel<<<TOK, 256, 0, stream>>>(bE, ln1_g, ln1_b, bL, DMODEL, 0);
    launch_gemm(stream, bL, cWq, bq, nullptr, nullptr, bQb, TOK, DMODEL, DMODEL, DMODEL, DMODEL, DMODEL, 0, scale);
    launch_gemm(stream, bL, cWk, nullptr, nullptr, nullptr, bKb, TOK, DMODEL, DMODEL, DMODEL, DMODEL, DMODEL, 0, 1.f);
    launch_gemm(stream, bL, cWv, bv, nullptr, nullptr, bVb, TOK, DMODEL, DMODEL, DMODEL, DMODEL, DMODEL, 0, 1.f);
    // S = Q @ K^T   (batch per (bt,h): base = (bh/H)*768 + (bh%H)*64)
    launch_gemm(stream, bQb, bKb, nullptr, nullptr, bS, nullptr, TN, TN, HDIM,
                LDQ, LDQ, TN, /*transB=*/1, 1.f, BH,
                NHEAD, DMODEL, HDIM, NHEAD, DMODEL, HDIM, 1, (long)NN, 0);
    k_softmax<<<BH * TN, 256, 0, stream>>>(bS, bBias, bPb);
    // attn = P @ V
    launch_gemm(stream, bPb, bVb, nullptr, nullptr, bA, nullptr, TN, HDIM, TN,
                TN, LDQ, LDQ, 0, 1.f, BH,
                1, (long)NN, 0, NHEAD, DMODEL, HDIM, NHEAD, DMODEL, HDIM);
    ln_kernel<<<TOK, 256, 0, stream>>>(bA, aln_g, aln_b, bL, DMODEL, 0);
    launch_gemm(stream, bL, cWo, bo, nullptr, bB, nullptr, TOK, DMODEL, DMODEL, DMODEL, DMODEL, DMODEL, 0, 1.f);  // h0
    k_cvt<<<eblocks(TOKD), 256, 0, stream>>>(bB, bH0b, TOKD);
    launch_gemm(stream, bH0b, cSa1, sa1b, nullptr, nullptr, bAdb, TOK, AD, DMODEL, DMODEL, AD, AD, 0, 1.f);
    k_gelu_bf<<<eblocks((long)TOK * AD), 256, 0, stream>>>(bAdb, (long)TOK * AD);
    launch_gemm(stream, bAdb, cSa2, sa2b, bB, bC, nullptr, TOK, DMODEL, AD, AD, DMODEL, DMODEL, 0, 1.f); // h = h0 + adapter
    k_axpg<<<eblocks(TOKD), 256, 0, stream>>>(x, gamma1, bC, bF);   // x2 = x + gamma1*h

    // ===== GLU FFN + adapter =====
    ln_kernel<<<TOK, 256, 0, stream>>>(bF, lnf_g, lnf_b, bX, DMODEL, 0);   // xn
    launch_gemm(stream, bX, cW0, nullptr, nullptr, bG0, nullptr, TOK, FFND, DMODEL, DMODEL, FFND, FFND, 0, 1.f);
    launch_gemm(stream, bX, cW1, nullptr, nullptr, bG1, nullptr, TOK, FFND, DMODEL, DMODEL, FFND, FFND, 0, 1.f);
    k_gelu_mul<<<eblocks((long)TOK * FFND), 256, 0, stream>>>(bG0, bG1, (long)TOK * FFND);
    ln_kernel<<<TOK, 256, 0, stream>>>(bG0, fln_g, fln_b, bGbf, FFND, 0);
    launch_gemm(stream, bGbf, cWf, bfb, nullptr, bB, nullptr, TOK, DMODEL, FFND, FFND, DMODEL, DMODEL, 0, 1.f);
    launch_gemm(stream, bX, cMa1, ma1b, nullptr, nullptr, bAdb, TOK, AD, DMODEL, DMODEL, AD, AD, 0, 1.f);
    k_gelu_bf<<<eblocks((long)TOK * AD), 256, 0, stream>>>(bAdb, (long)TOK * AD);
    launch_gemm(stream, bAdb, cMa2, ma2b, nullptr, bC, nullptr, TOK, DMODEL, AD, AD, DMODEL, DMODEL, 0, 1.f);
    k_final<<<eblocks(TOKD), 256, 0, stream>>>(bF, gamma2, bB, bC, (float*)d_out);
}
